// DualGraphEncoder_43928925503608
// MI455X (gfx1250) — matrix operand order repaired and checked by reading, not compiled, not run on hardware
//
#include <hip/hip_runtime.h>
#include <math.h>

typedef __attribute__((ext_vector_type(16))) __bf16 v16bf;
typedef __attribute__((ext_vector_type(8)))  __bf16 v8bf;
typedef __attribute__((ext_vector_type(8)))  float  v8f;

#define NNODE 4096
#define NEDGE 131072

// ---------------- block reductions ----------------
__device__ __forceinline__ float bsum(float v, float* red, int T) {
  int t = threadIdx.x;
  red[t] = v; __syncthreads();
  for (int s = T >> 1; s > 0; s >>= 1) { if (t < s) red[t] += red[t + s]; __syncthreads(); }
  float r = red[0]; __syncthreads(); return r;
}
__device__ __forceinline__ float bmaxr(float v, float* red, int T) {
  int t = threadIdx.x;
  red[t] = v; __syncthreads();
  for (int s = T >> 1; s > 0; s >>= 1) { if (t < s) red[t] = fmaxf(red[t], red[t + s]); __syncthreads(); }
  float r = red[0]; __syncthreads(); return r;
}

// ---------------- elementwise helpers ----------------
__global__ void k_cvt_bf16(const float* __restrict__ s, __bf16* __restrict__ d, int n) {
  int i = blockIdx.x * blockDim.x + threadIdx.x;
  if (i < n) d[i] = (__bf16)s[i];
}

__global__ void k_cnt(const int* __restrict__ rows, float* __restrict__ cnt, int E) {
  int i = blockIdx.x * blockDim.x + threadIdx.x;
  if (i < E) atomicAdd(&cnt[rows[i]], 1.0f);
}

__global__ void k_scatter(const int* __restrict__ rows, const int* __restrict__ cols,
                          const float* __restrict__ x, float* __restrict__ acc, int E, int D) {
  int i = blockIdx.x * blockDim.x + threadIdx.x;
  if (i >= E * D) return;
  int e = i / D, d = i - e * D;
  atomicAdd(&acc[(size_t)rows[e] * D + d], x[(size_t)cols[e] * D + d]);
}

__global__ void k_mean(const float* __restrict__ acc, const float* __restrict__ cnt,
                       __bf16* __restrict__ out, int n, int D) {
  int i = blockIdx.x * blockDim.x + threadIdx.x;
  if (i >= n * D) return;
  out[i] = (__bf16)(acc[i] / (cnt[i / D] + 1e-12f));
}

// LayerNorm (+ optional exact GELU), one block per row, blockDim == D (128 or 256)
__global__ void k_ln(const float* __restrict__ in, const float* __restrict__ g,
                     const float* __restrict__ b, float* __restrict__ outf,
                     __bf16* __restrict__ outb, int D, int dogelu) {
  __shared__ float red[256];
  int n = blockIdx.x, t = threadIdx.x;
  float v = in[(size_t)n * D + t];
  float mu = bsum(v, red, blockDim.x) / (float)D;
  float dv = v - mu;
  float var = bsum(dv * dv, red, blockDim.x) / (float)D;
  float y = dv * rsqrtf(var + 1e-5f) * g[t] + b[t];
  if (dogelu) y = 0.5f * y * (1.0f + erff(y * 0.70710678118f));
  if (outf) outf[(size_t)n * D + t] = y;
  if (outb) outb[(size_t)n * D + t] = (__bf16)y;
}

// LN over the 768-wide concat [h1s|h2s|h1a|h2a] -> bf16
__global__ void k_concat_ln(const float* __restrict__ h1s, const float* __restrict__ h2s,
                            const float* __restrict__ h1a, const float* __restrict__ h2a,
                            const float* __restrict__ g, const float* __restrict__ b,
                            __bf16* __restrict__ out) {
  __shared__ float red[256];
  int n = blockIdx.x, t = threadIdx.x;
  float vals[3]; int cols[3];
  float ls = 0.f;
  for (int i = 0; i < 3; ++i) {
    int j = t + i * 256; cols[i] = j;
    float v;
    if (j < 256)       v = h1s[(size_t)n * 256 + j];
    else if (j < 384)  v = h2s[(size_t)n * 128 + (j - 256)];
    else if (j < 640)  v = h1a[(size_t)n * 256 + (j - 384)];
    else               v = h2a[(size_t)n * 128 + (j - 640)];
    vals[i] = v; ls += v;
  }
  float mu = bsum(ls, red, 256) / 768.f;
  float lv = 0.f;
  for (int i = 0; i < 3; ++i) { float d = vals[i] - mu; lv += d * d; }
  float var = bsum(lv, red, 256) / 768.f;
  float inv = rsqrtf(var + 1e-5f);
  for (int i = 0; i < 3; ++i)
    out[(size_t)n * 768 + cols[i]] = (__bf16)((vals[i] - mu) * inv * g[cols[i]] + b[cols[i]]);
}

// w = sigmoid(g . W2 + b2); fused = w*hs + (1-w)*ha  (bf16 out)
__global__ void k_gate_fuse(const float* __restrict__ gbuf, const float* __restrict__ W2,
                            const float* __restrict__ b2,
                            const float* __restrict__ h1s, const float* __restrict__ h2s,
                            const float* __restrict__ h1a, const float* __restrict__ h2a,
                            __bf16* __restrict__ fbuf) {
  __shared__ float red[256];
  int n = blockIdx.x, t = threadIdx.x;
  float v = (t < 128) ? gbuf[(size_t)n * 128 + t] * W2[t] : 0.f;
  float s = bsum(v, red, 256);
  float w = 1.f / (1.f + __expf(-(s + b2[0])));
  for (int j = t; j < 384; j += 256) {
    float hs, ha;
    if (j < 256) { hs = h1s[(size_t)n * 256 + j]; ha = h1a[(size_t)n * 256 + j]; }
    else { hs = h2s[(size_t)n * 128 + (j - 256)]; ha = h2a[(size_t)n * 128 + (j - 256)]; }
    fbuf[(size_t)n * 384 + j] = (__bf16)(w * hs + (1.f - w) * ha);
  }
}

// Qg[g][n][k] = q[n][k] * Wpre[g][k/32] * 1/sqrt(DK)   (folds pre-mix + scale into QK^T)
__global__ void k_scale_q(const __bf16* __restrict__ qb, const float* __restrict__ Wpre,
                          __bf16* __restrict__ qg, int n) {
  int i = blockIdx.x * blockDim.x + threadIdx.x;
  if (i >= 4 * n * 128) return;
  int k = i & 127; int rem = i >> 7; int node = rem % n; int g = rem / n;
  float v = (float)qb[(size_t)node * 128 + k] * Wpre[g * 4 + (k >> 5)] * 0.17677669529663687f;
  qg[i] = (__bf16)v;
}

// per-(head,row): softmax -> bisection for 128th-largest -> sparsify+renorm, in place (bf16)
__global__ void k_softmax_topk(__bf16* __restrict__ sc, int n) {
  __shared__ float rbuf[NNODE];
  __shared__ float red[256];
  int row = blockIdx.x, h = blockIdx.y, t = threadIdx.x;
  const int T = 256;
  __bf16* p = sc + (size_t)h * n * n + (size_t)row * n;
  for (int i = t; i < n; i += T) rbuf[i] = (float)p[i];
  __syncthreads();
  float lm = -3.4e38f;
  for (int i = t; i < n; i += T) lm = fmaxf(lm, rbuf[i]);
  float m = bmaxr(lm, red, T);
  float ls = 0.f;
  for (int i = t; i < n; i += T) { float e = __expf(rbuf[i] - m); rbuf[i] = e; ls += e; }
  float s = bsum(ls, red, T);
  float inv = 1.f / s;
  for (int i = t; i < n; i += T) rbuf[i] *= inv;
  __syncthreads();
  float lo = 0.f, hi = 1.f;
  for (int it = 0; it < 30; ++it) {
    float mid = 0.5f * (lo + hi);
    float lc = 0.f;
    for (int i = t; i < n; i += T) lc += (rbuf[i] >= mid) ? 1.f : 0.f;
    float c = bsum(lc, red, T);
    if (c >= 128.f) lo = mid; else hi = mid;
  }
  float thr = lo, ls2 = 0.f;
  for (int i = t; i < n; i += T) { float v = (rbuf[i] >= thr) ? rbuf[i] : 0.f; rbuf[i] = v; ls2 += v; }
  float s2 = bsum(ls2, red, T);
  float inv2 = 1.f / (s2 + 1e-9f);
  for (int i = t; i < n; i += T) p[i] = (__bf16)(rbuf[i] * inv2);
}

// post-softmax head mixing, in place (each (n,i) reads all 4 planes before writing)
__global__ void k_postmix(__bf16* __restrict__ sc, const float* __restrict__ Wpost,
                          const float* __restrict__ bpost, int n) {
  int row = blockIdx.x, t = threadIdx.x;
  size_t plane = (size_t)n * n;
  __bf16* base = sc + (size_t)row * n;
  for (int i = t; i < n; i += blockDim.x) {
    float a0 = (float)base[i];
    float a1 = (float)base[plane + i];
    float a2 = (float)base[2 * plane + i];
    float a3 = (float)base[3 * plane + i];
    float o0 = bpost[0] + Wpost[0] * a0 + Wpost[1] * a1 + Wpost[2] * a2 + Wpost[3] * a3;
    float o1 = bpost[1] + Wpost[4] * a0 + Wpost[5] * a1 + Wpost[6] * a2 + Wpost[7] * a3;
    float o2 = bpost[2] + Wpost[8] * a0 + Wpost[9] * a1 + Wpost[10] * a2 + Wpost[11] * a3;
    float o3 = bpost[3] + Wpost[12] * a0 + Wpost[13] * a1 + Wpost[14] * a2 + Wpost[15] * a3;
    base[i] = (__bf16)o0;
    base[plane + i] = (__bf16)o1;
    base[2 * plane + i] = (__bf16)o2;
    base[3 * plane + i] = (__bf16)o3;
  }
}

__global__ void k_normalize(const float* __restrict__ z, float* __restrict__ out) {
  __shared__ float red[128];
  int n = blockIdx.x, t = threadIdx.x;
  float v = z[(size_t)n * 128 + t];
  float ss = bsum(v * v, red, 128);
  float nrm = fmaxf(sqrtf(ss), 1e-12f);
  out[(size_t)n * 128 + t] = v / nrm;
}

// ---------------- generic WMMA bf16 GEMM ----------------
// C = act( A*B (+A2*B2) + bias (+bias2) + sbias[sidx] + resid ), f32/bf16 out.
// 256 threads = 8 wave32 waves; 128x64 C tile; wave = 32x32 (2x2 WMMA frags); BK=32.
// Requires M % 128 == 0 and K % 32 == 0 (true for every call site).
// LDS staging is "fragment-ready":
//  - ldsA rows hold [K0..31] in natural order so each lane's v16bf A
//    fragment (16 consecutive K at hset*16) is one contiguous 32B LDS read.
//  - ldsBT is n-major [n][k]; a lane's B fragment (16 consecutive K at hset*16)
//    is likewise one contiguous 32B read.
__global__ __launch_bounds__(256) void k_gemm(
    const __bf16* __restrict__ A, const __bf16* __restrict__ B,
    const __bf16* __restrict__ A2, const __bf16* __restrict__ B2,
    const float* __restrict__ bias, const float* __restrict__ bias2,
    const float* __restrict__ sbias, int sidx,
    const float* __restrict__ resid,
    float* __restrict__ Cf, __bf16* __restrict__ Cb,
    int M, int Ntot, int K, int K2,
    int lda, int ldb, int lda2, int ldb2,
    int ldc, int transB, int act) {
  __shared__ alignas(16) __bf16 ldsA[128][40];
  __shared__ alignas(16) __bf16 ldsBT[64][40];

  const int tid = threadIdx.x;
  const int wave = tid >> 5;
  const int lane = tid & 31;
  const int hset = lane >> 4;   // which 16-lane half
  const int l    = lane & 15;
  const int wm = wave >> 1;     // 0..3
  const int wn = wave & 1;      // 0..1
  const int m0 = blockIdx.y * 128;
  const int n0 = blockIdx.x * 64;

  v8f acc[2][2];
  for (int mi = 0; mi < 2; ++mi)
    for (int ni = 0; ni < 2; ++ni)
      for (int e = 0; e < 8; ++e) acc[mi][ni][e] = 0.f;

  for (int pass = 0; pass < 2; ++pass) {
    const __bf16* Ap = pass ? A2 : A;
    const __bf16* Bp = pass ? B2 : B;
    const int Kp   = pass ? K2 : K;
    const int ldap = pass ? lda2 : lda;
    const int ldbp = pass ? ldb2 : ldb;
    if (Ap == nullptr || Kp <= 0) continue;

    for (int k0 = 0; k0 < Kp; k0 += 32) {
      // ---- A tile 128x32, natural chunk store (no swizzle) ----
      for (int it = 0; it < 2; ++it) {
        int c = tid + it * 256;          // 0..511
        int row = c >> 2;
        int col = (c & 3) << 3;          // 0,8,16,24
        v8bf v = *(const v8bf*)(Ap + (size_t)(m0 + row) * ldap + k0 + col);
        int scol = col;
        *(v8bf*)(&ldsA[row][scol]) = v;
      }
      // prefetch next A K-tile (lowers to global_prefetch_b8)
      if (k0 + 32 < Kp)
        __builtin_prefetch(Ap + (size_t)(m0 + (tid >> 2)) * ldap + k0 + 32, 0, 1);

      // ---- B tile 32x64 -> ldsBT[n][k] (n-major) ----
      if (!transB) {
        // B global [K, Ntot] row-major: coalesced 8-n vector load, scatter into BT
        int k  = tid >> 3;
        int nc = (tid & 7) << 3;
        int gn = n0 + nc;
        v8bf v;
        if (gn + 8 <= Ntot) {
          v = *(const v8bf*)(Bp + (size_t)(k0 + k) * ldbp + gn);
        } else {
          for (int j = 0; j < 8; ++j)
            v[j] = (gn + j < Ntot) ? Bp[(size_t)(k0 + k) * ldbp + gn + j] : (__bf16)0.f;
        }
        for (int j = 0; j < 8; ++j) ldsBT[nc + j][k] = v[j];
      } else {
        // B global [Ntot, K] row-major: vector load 8 consecutive k, vector store
        int nn = tid >> 2;
        int kc = (tid & 3) << 3;
        int gn = n0 + nn;
        v8bf v;
        if (gn < Ntot) v = *(const v8bf*)(Bp + (size_t)gn * ldbp + k0 + kc);
        else           for (int j = 0; j < 8; ++j) v[j] = (__bf16)0.f;
        *(v8bf*)(&ldsBT[nn][kc]) = v;
      }
      __syncthreads();

      // ---- fragments: single contiguous v16bf LDS reads ----
      v16bf afr[2], bfr[2];
      for (int mi = 0; mi < 2; ++mi)
        afr[mi] = *(const v16bf*)(&ldsA[wm * 32 + mi * 16 + l][hset * 16]);
      for (int ni = 0; ni < 2; ++ni)
        bfr[ni] = *(const v16bf*)(&ldsBT[wn * 32 + ni * 16 + l][hset * 16]);

      for (int mi = 0; mi < 2; ++mi)
        for (int ni = 0; ni < 2; ++ni)
          acc[mi][ni] = __builtin_amdgcn_wmma_f32_16x16x32_bf16(
              false, afr[mi], false, bfr[ni], (short)0, acc[mi][ni], false, false);
      __syncthreads();
    }
  }

  // ---- epilogue (C layout: lane half selects M block of 8; l = column) ----
  for (int mi = 0; mi < 2; ++mi) {
    for (int ni = 0; ni < 2; ++ni) {
      for (int r = 0; r < 8; ++r) {
        int gm = m0 + wm * 32 + mi * 16 + hset * 8 + r;
        int gn = n0 + wn * 32 + ni * 16 + l;
        if (gn < Ntot) {
          float v = acc[mi][ni][r];
          if (bias)  v += bias[gn];
          if (bias2) v += bias2[gn];
          if (sbias) v += sbias[sidx];
          if (resid) v += resid[(size_t)gm * ldc + gn];
          if (act == 1) v = fmaxf(v, 0.f);
          if (Cf) Cf[(size_t)gm * ldc + gn] = v;
          if (Cb) Cb[(size_t)gm * ldc + gn] = (__bf16)v;
        }
      }
    }
  }
}

// ---------------- host launcher ----------------
extern "C" void kernel_launch(void* const* d_in, const int* in_sizes, int n_in,
                              void* d_out, int out_size, void* d_ws, size_t ws_size,
                              hipStream_t stream) {
  (void)in_sizes; (void)n_in; (void)out_size; (void)ws_size;
  const int N = NNODE, E = NEDGE;
  auto P = [&](int i) { return (const float*)d_in[i]; };
  const float* x   = P(0);
  const int* esp   = (const int*)d_in[1];   // [2,E]: rows then cols
  const int* eat   = (const int*)d_in[2];

  // params (insertion-order flattening of setup_inputs)
  // s[0]:3..8  s[1]:9..14  a[0]:15..20  a[1]:21..26
  // nc_g=27 nc_b=28 fg_W1=29 fg_b1=30 fg_W2=31 fg_b2=32 rp_W=33 rp_b=34
  // ln_g=35 ln_b=36 Wq=37 bq=38 Wk=39 bk=40 Wv=41 bv=42
  // Wpre=43 bpre=44 Wpost=45 bpost=46 Wo=47 bo=48

  char* W = (char*)d_ws;
  size_t off = 0;
  auto alloc = [&](size_t bytes) { size_t o = off; off += (bytes + 255) & ~(size_t)255; return o; };

  __bf16* xb      = (__bf16*)(W + alloc((size_t)N * 256 * 2));
  __bf16* wb_s0Ws = (__bf16*)(W + alloc(256 * 256 * 2));
  __bf16* wb_s0Wn = (__bf16*)(W + alloc(256 * 256 * 2));
  __bf16* wb_s1Ws = (__bf16*)(W + alloc(256 * 128 * 2));
  __bf16* wb_s1Wn = (__bf16*)(W + alloc(256 * 128 * 2));
  __bf16* wb_a0Ws = (__bf16*)(W + alloc(256 * 256 * 2));
  __bf16* wb_a0Wn = (__bf16*)(W + alloc(256 * 256 * 2));
  __bf16* wb_a1Ws = (__bf16*)(W + alloc(256 * 128 * 2));
  __bf16* wb_a1Wn = (__bf16*)(W + alloc(256 * 128 * 2));
  __bf16* wb_fgW1 = (__bf16*)(W + alloc(768 * 128 * 2));
  __bf16* wb_rpW  = (__bf16*)(W + alloc(384 * 128 * 2));
  __bf16* wb_Wq   = (__bf16*)(W + alloc(128 * 128 * 2));
  __bf16* wb_Wk   = (__bf16*)(W + alloc(128 * 128 * 2));
  __bf16* wb_Wv   = (__bf16*)(W + alloc(128 * 128 * 2));
  __bf16* wb_Wo   = (__bf16*)(W + alloc(128 * 128 * 2));
  float*  cnt_s   = (float*)(W + alloc((size_t)N * 4));
  float*  cnt_a   = (float*)(W + alloc((size_t)N * 4));
  float*  accb    = (float*)(W + alloc((size_t)N * 256 * 4));
  __bf16* neib    = (__bf16*)(W + alloc((size_t)N * 256 * 2));
  float*  gtmp    = (float*)(W + alloc((size_t)N * 256 * 4));
  float*  h1s     = (float*)(W + alloc((size_t)N * 256 * 4));
  __bf16* h1sb    = (__bf16*)(W + alloc((size_t)N * 256 * 2));
  float*  h2s     = (float*)(W + alloc((size_t)N * 128 * 4));
  float*  h1a     = (float*)(W + alloc((size_t)N * 256 * 4));
  __bf16* h1ab    = (__bf16*)(W + alloc((size_t)N * 256 * 2));
  float*  h2a     = (float*)(W + alloc((size_t)N * 128 * 4));
  __bf16* cbuf    = (__bf16*)(W + alloc((size_t)N * 768 * 2));
  float*  gbuf    = (float*)(W + alloc((size_t)N * 128 * 4));
  __bf16* fbuf    = (__bf16*)(W + alloc((size_t)N * 384 * 2));
  float*  zres    = (float*)(W + alloc((size_t)N * 128 * 4));
  __bf16* xnb     = (__bf16*)(W + alloc((size_t)N * 128 * 2));
  __bf16* qb      = (__bf16*)(W + alloc((size_t)N * 128 * 2));
  __bf16* kb      = (__bf16*)(W + alloc((size_t)N * 128 * 2));
  __bf16* vb      = (__bf16*)(W + alloc((size_t)N * 128 * 2));
  __bf16* qg      = (__bf16*)(W + alloc((size_t)4 * N * 128 * 2));
  __bf16* scores  = (__bf16*)(W + alloc((size_t)4 * N * N * 2));  // in-place attn after softmax
  float*  obuf    = (float*)(W + alloc((size_t)N * 128 * 4));
  __bf16* obb     = (__bf16*)(W + alloc((size_t)N * 128 * 2));
  float*  z2      = (float*)(W + alloc((size_t)N * 128 * 4));

  auto cvt = [&](const float* src, __bf16* dst, int n) {
    k_cvt_bf16<<<(n + 255) / 256, 256, 0, stream>>>(src, dst, n);
  };
  auto gemm = [&](const __bf16* A, const __bf16* B, const __bf16* A2, const __bf16* B2,
                  const float* bias, const float* bias2, const float* sb, int sidx,
                  const float* resid, float* Cf, __bf16* Cb,
                  int M, int Nt, int K, int K2, int lda, int ldb, int lda2, int ldb2,
                  int ldc, int transB, int act) {
    dim3 grid((Nt + 63) / 64, (M + 127) / 128);
    k_gemm<<<grid, 256, 0, stream>>>(A, B, A2, B2, bias, bias2, sb, sidx, resid, Cf, Cb,
                                     M, Nt, K, K2, lda, ldb, lda2, ldb2, ldc, transB, act);
  };

  // ---- bf16 conversions ----
  cvt(x, xb, N * 256);
  cvt(P(3),  wb_s0Ws, 256 * 256); cvt(P(5),  wb_s0Wn, 256 * 256);
  cvt(P(9),  wb_s1Ws, 256 * 128); cvt(P(11), wb_s1Wn, 256 * 128);
  cvt(P(15), wb_a0Ws, 256 * 256); cvt(P(17), wb_a0Wn, 256 * 256);
  cvt(P(21), wb_a1Ws, 256 * 128); cvt(P(23), wb_a1Wn, 256 * 128);
  cvt(P(29), wb_fgW1, 768 * 128);
  cvt(P(33), wb_rpW,  384 * 128);
  cvt(P(37), wb_Wq, 128 * 128); cvt(P(39), wb_Wk, 128 * 128);
  cvt(P(41), wb_Wv, 128 * 128); cvt(P(47), wb_Wo, 128 * 128);

  // ---- degree counts (fixed per stream) ----
  hipMemsetAsync(cnt_s, 0, (size_t)N * 4, stream);
  hipMemsetAsync(cnt_a, 0, (size_t)N * 4, stream);
  k_cnt<<<(E + 255) / 256, 256, 0, stream>>>(esp, cnt_s, E);
  k_cnt<<<(E + 255) / 256, 256, 0, stream>>>(eat, cnt_a, E);

  const int scblk = (E * 256) / 256;  // E*D threads / 256
  // ---- structural stream ----
  hipMemsetAsync(accb, 0, (size_t)N * 256 * 4, stream);
  k_scatter<<<scblk, 256, 0, stream>>>(esp, esp + E, x, accb, E, 256);
  k_mean<<<(N * 256 + 255) / 256, 256, 0, stream>>>(accb, cnt_s, neib, N, 256);
  gemm(xb, wb_s0Ws, neib, wb_s0Wn, P(4), P(6), nullptr, 0, nullptr, gtmp, nullptr,
       N, 256, 256, 256, 256, 256, 256, 256, 256, 0, 0);
  k_ln<<<N, 256, 0, stream>>>(gtmp, P(7), P(8), h1s, h1sb, 256, 1);
  hipMemsetAsync(accb, 0, (size_t)N * 256 * 4, stream);
  k_scatter<<<scblk, 256, 0, stream>>>(esp, esp + E, h1s, accb, E, 256);
  k_mean<<<(N * 256 + 255) / 256, 256, 0, stream>>>(accb, cnt_s, neib, N, 256);
  gemm(h1sb, wb_s1Ws, neib, wb_s1Wn, P(10), P(12), nullptr, 0, nullptr, gtmp, nullptr,
       N, 128, 256, 256, 256, 128, 256, 128, 128, 0, 0);
  k_ln<<<N, 128, 0, stream>>>(gtmp, P(13), P(14), h2s, nullptr, 128, 1);

  // ---- attribute stream ----
  hipMemsetAsync(accb, 0, (size_t)N * 256 * 4, stream);
  k_scatter<<<scblk, 256, 0, stream>>>(eat, eat + E, x, accb, E, 256);
  k_mean<<<(N * 256 + 255) / 256, 256, 0, stream>>>(accb, cnt_a, neib, N, 256);
  gemm(xb, wb_a0Ws, neib, wb_a0Wn, P(16), P(18), nullptr, 0, nullptr, gtmp, nullptr,
       N, 256, 256, 256, 256, 256, 256, 256, 256, 0, 0);
  k_ln<<<N, 256, 0, stream>>>(gtmp, P(19), P(20), h1a, h1ab, 256, 1);
  hipMemsetAsync(accb, 0, (size_t)N * 256 * 4, stream);
  k_scatter<<<scblk, 256, 0, stream>>>(eat, eat + E, h1a, accb, E, 256);
  k_mean<<<(N * 256 + 255) / 256, 256, 0, stream>>>(accb, cnt_a, neib, N, 256);
  gemm(h1ab, wb_a1Ws, neib, wb_a1Wn, P(22), P(24), nullptr, 0, nullptr, gtmp, nullptr,
       N, 128, 256, 256, 256, 128, 256, 128, 128, 0, 0);
  k_ln<<<N, 128, 0, stream>>>(gtmp, P(25), P(26), h2a, nullptr, 128, 1);

  // ---- fusion gate + reprojection ----
  k_concat_ln<<<N, 256, 0, stream>>>(h1s, h2s, h1a, h2a, P(27), P(28), cbuf);
  gemm(cbuf, wb_fgW1, nullptr, nullptr, P(30), nullptr, nullptr, 0, nullptr, gbuf, nullptr,
       N, 128, 768, 0, 768, 128, 0, 0, 128, 0, 1);
  k_gate_fuse<<<N, 256, 0, stream>>>(gbuf, P(31), P(32), h1s, h2s, h1a, h2a, fbuf);
  gemm(fbuf, wb_rpW, nullptr, nullptr, P(34), nullptr, nullptr, 0, nullptr, zres, nullptr,
       N, 128, 384, 0, 384, 128, 0, 0, 128, 0, 0);

  // ---- attention ----
  k_ln<<<N, 128, 0, stream>>>(zres, P(35), P(36), nullptr, xnb, 128, 0);
  gemm(xnb, wb_Wq, nullptr, nullptr, P(38), nullptr, nullptr, 0, nullptr, nullptr, qb,
       N, 128, 128, 0, 128, 128, 0, 0, 128, 0, 0);
  gemm(xnb, wb_Wk, nullptr, nullptr, P(40), nullptr, nullptr, 0, nullptr, nullptr, kb,
       N, 128, 128, 0, 128, 128, 0, 0, 128, 0, 0);
  gemm(xnb, wb_Wv, nullptr, nullptr, P(42), nullptr, nullptr, 0, nullptr, nullptr, vb,
       N, 128, 128, 0, 128, 128, 0, 0, 128, 0, 0);
  k_scale_q<<<(4 * N * 128 + 255) / 256, 256, 0, stream>>>(qb, P(43), qg, N);

  size_t plane = (size_t)N * N;
  for (int g = 0; g < 4; ++g)  // mixed scores[g] = Qg[g] @ K^T + bpre[g]
    gemm(qg + (size_t)g * N * 128, kb, nullptr, nullptr, nullptr, nullptr, P(44), g,
         nullptr, nullptr, scores + (size_t)g * plane,
         N, N, 128, 0, 128, 128, 0, 0, N, 1, 0);

  dim3 smgrid(N, 4);
  k_softmax_topk<<<smgrid, 256, 0, stream>>>(scores, N);
  k_postmix<<<N, 256, 0, stream>>>(scores, P(45), P(46), N);

  for (int g = 0; g < 4; ++g)  // out[:, g*32:(g+1)*32] = attn[g] @ V_g
    gemm(scores + (size_t)g * plane, vb + g * 32, nullptr, nullptr, nullptr, nullptr,
         nullptr, 0, nullptr, obuf + g * 32, nullptr,
         N, 32, N, 0, N, 128, 0, 0, 128, 0, 0);

  cvt(obuf, obb, N * 128);
  gemm(obb, wb_Wo, nullptr, nullptr, P(48), nullptr, nullptr, 0, zres, z2, nullptr,
       N, 128, 128, 0, 128, 128, 0, 0, 128, 0, 0);
  k_normalize<<<N, 128, 0, stream>>>(z2, (float*)d_out);
}
